// MultiHeadAttention_90142773609184
// MI455X (gfx1250) — matrix operand order repaired and checked by reading, not compiled, not run on hardware
//
#include <hip/hip_runtime.h>

// ---------------- CDNA5 / gfx1250 types ----------------
typedef __attribute__((ext_vector_type(16))) __bf16        v16bf;
typedef __attribute__((ext_vector_type(8)))  float         v8f;
typedef __attribute__((ext_vector_type(4)))  unsigned int  v4u;
typedef __attribute__((ext_vector_type(4)))  int           v4i;

#define BQ   4
#define SQ   2048
#define DQ   1024
#define HQ   16
#define HDQ  64
#define MQ   (BQ * SQ)      // 8192 rows

struct U256 { v4u a, b; };

// Load a 16-element bf16 fragment as two 128-bit loads (p0 -> elems 0..7, p1 -> 8..15).
__device__ __forceinline__ v16bf load_frag2(const __bf16* p0, const __bf16* p1) {
    U256 t;
    t.a = *reinterpret_cast<const v4u*>(p0);
    t.b = *reinterpret_cast<const v4u*>(p1);
    return __builtin_bit_cast(v16bf, t);
}

__device__ __forceinline__ v8f wmma_bf16(v16bf a, v16bf b, v8f c) {
    return __builtin_amdgcn_wmma_f32_16x16x32_bf16(false, a, false, b, (short)0, c,
                                                   false, false);
}

// ---------------- async global -> LDS copy (CDNA5 path, guarded) ----------------
#if __has_builtin(__builtin_amdgcn_global_load_async_to_lds_b128)
#define HAVE_ASYNC_LDS 1
#else
#define HAVE_ASYNC_LDS 0
#endif

__device__ __forceinline__ void async_copy16(const __bf16* g, __bf16* l) {
#if HAVE_ASYNC_LDS
    __builtin_amdgcn_global_load_async_to_lds_b128(
        (__attribute__((address_space(1))) v4i*)(v4i*)const_cast<__bf16*>(g),
        (__attribute__((address_space(3))) v4i*)(v4i*)l, 0, 0);
#else
    *reinterpret_cast<v4u*>(l) = *reinterpret_cast<const v4u*>(g);
#endif
}

__device__ __forceinline__ void async_wait0() {
#if HAVE_ASYNC_LDS
#if __has_builtin(__builtin_amdgcn_s_wait_asynccnt)
    __builtin_amdgcn_s_wait_asynccnt(0);
#else
    asm volatile("s_wait_asynccnt 0" ::: "memory");
#endif
#endif
}

// ---------------- fp32 -> bf16 cast ----------------
__global__ __launch_bounds__(256) void cvt_f32_bf16(const float* __restrict__ in,
                                                    __bf16* __restrict__ out, int n) {
    int i = blockIdx.x * 256 + threadIdx.x;
    if (i < n) out[i] = (__bf16)in[i];
}

// ---------------- weight transpose + cast: wT[n*D + k] = w[k*D + n] ----------------
__global__ __launch_bounds__(256) void transpose_cvt(const float* __restrict__ w,
                                                     __bf16* __restrict__ wT) {
    int k = blockIdx.x * 256 + threadIdx.x;   // coalesced store along k
    int n = blockIdx.y;
    wT[(size_t)n * DQ + k] = (__bf16)w[(size_t)k * DQ + n];
}

// ---------------- WMMA GEMM: C[M,N] = A[M,K] * BT[N,K]^T ----------------
// B tile (64 x 32 k-slice, shared by all 8 waves) is double-buffered in LDS via
// async global->LDS copies (ASYNCcnt); A fragments stream from global per wave.
// MODE 0: fp32 row-major out.  MODE 1: bf16 out to [B,H,S,HD].  MODE 2: bf16 out to [B,H,HD,S].
// `scale` is folded into the epilogue (used to pre-scale Q by 1/sqrt(HD); exact in bf16).
template <int MODE>
__global__ __launch_bounds__(256)
void gemm_wmma_bf16(const __bf16* __restrict__ A, const __bf16* __restrict__ BT,
                    float* __restrict__ outF, __bf16* __restrict__ outB,
                    int M, int N, int K, float scale) {
    __shared__ alignas(16) __bf16 bsh[2][64 * 32];   // 2 x 4 KB

    const int tid  = threadIdx.x;
    const int lane = tid & 31;
    const int wave = tid >> 5;
    const int lo   = lane & 15;
    const int hi   = lane >> 4;
    const int rowBase = blockIdx.y * 128 + wave * 16;
    const int colBase = blockIdx.x * 64;

    // copy mapping: thread t moves 16 B: BT row n = t>>2, 8-elem chunk c = t&3
    const int cn = tid >> 2, cc = tid & 3;
    const __bf16* bSrc = BT + (size_t)(colBase + cn) * K + cc * 8;
    __bf16* bDst0 = &bsh[0][cn * 32 + cc * 8];
    __bf16* bDst1 = &bsh[1][cn * 32 + cc * 8];

    async_copy16(bSrc, bDst0);          // preload k-slice 0
    async_wait0();
    __syncthreads();

    v8f acc[4] = {};
    const __bf16* aRow = A + (size_t)(rowBase + lo) * K;

    const int nk = K / 32;
    for (int ik = 0; ik < nk; ++ik) {
        const int k0  = ik * 32;
        const int buf = ik & 1;
        if (ik + 1 < nk)                 // prefetch next k-slice into the other buffer
            async_copy16(bSrc + (size_t)(ik + 1) * 32, (buf ? bDst0 : bDst1));

        // A fragment 16x32: lane row = lo, K = hi*16..+15
        v16bf af = load_frag2(aRow + k0 + hi * 16, aRow + k0 + hi * 16 + 8);
#pragma unroll
        for (int t = 0; t < 4; ++t) {
            // B fragment 32x16 from LDS: column n = t*16+lo, K = hi*16..+15
            const __bf16* bp = &bsh[buf][(t * 16 + lo) * 32 + hi * 16];
            v16bf bfr = load_frag2(bp, bp + 8);
            acc[t] = wmma_bf16(af, bfr, acc[t]);
        }
        async_wait0();
        __syncthreads();                 // uniform trip count across all 8 waves
    }

#pragma unroll
    for (int t = 0; t < 4; ++t) {
#pragma unroll
        for (int j = 0; j < 8; ++j) {
            int row = rowBase + j + 8 * hi;          // C frag: row = j + 8*hi
            int col = colBase + t * 16 + lo;         //          col = lo
            float v = acc[t][j] * scale;
            if (MODE == 0) {
                outF[(size_t)row * N + col] = v;
            } else {
                int b = row >> 11, s = row & (SQ - 1);
                int h = col >> 6,  hd = col & (HDQ - 1);
                if (MODE == 1)   // [B,H,S,HD]
                    outB[(((size_t)(b * HQ + h)) * SQ + s) * HDQ + hd] = (__bf16)v;
                else             // [B,H,HD,S]  (V transposed)
                    outB[(((size_t)(b * HQ + h)) * HDQ + hd) * SQ + s] = (__bf16)v;
            }
        }
    }
}

// ---------------- one 64-key kv block of flash attention ----------------
// MASK is a call-site constant: only the final block of each q-tile crosses the
// diagonal (block ends are == 63 mod 64, q0 == 0 mod 16), so the steady-state
// loop is compiled entirely mask-free.
__device__ __forceinline__ void attn_kv_block(
    bool MASK, int kv0, int q0, int lo, int hi, int bh,
    const __bf16* __restrict__ Km, const __bf16* __restrict__ Vt, __bf16* pw,
    v16bf qa0, v16bf qa1, v16bf ones,
    v8f (&cacc)[4], float (&mi)[8], float (&li)[8]) {

    // ---- scores: four 16x16 tiles covering keys kv0..kv0+63, full HD=64 ----
    v8f s[4] = {};
#pragma unroll
    for (int tt = 0; tt < 4; ++tt) {
        const __bf16* kRow = Km + ((size_t)bh * SQ + kv0 + tt * 16 + lo) * HDQ;
        v16bf kb0 = load_frag2(kRow + hi * 16,      kRow + hi * 16 + 8);
        v16bf kb1 = load_frag2(kRow + 32 + hi * 16, kRow + 32 + hi * 16 + 8);
        s[tt] = wmma_bf16(qa0, kb0, s[tt]);
        s[tt] = wmma_bf16(qa1, kb1, s[tt]);
    }

    float alpha[8];
#pragma unroll
    for (int j = 0; j < 8; ++j) {
        if (MASK) {
            const int qg = q0 + j + 8 * hi;
#pragma unroll
            for (int tt = 0; tt < 4; ++tt)
                if (kv0 + tt * 16 + lo > qg) s[tt][j] = -3.0e38f;
        }
        float mb = fmaxf(fmaxf(s[0][j], s[1][j]), fmaxf(s[2][j], s[3][j]));
#pragma unroll
        for (int d = 1; d < 16; d <<= 1) mb = fmaxf(mb, __shfl_xor(mb, d, 32));
        float mnew = fmaxf(mi[j], mb);
        alpha[j] = __expf(mi[j] - mnew);
        mi[j] = mnew;
        const int r = (j + 8 * hi) * 64;
#pragma unroll
        for (int tt = 0; tt < 4; ++tt) {
            float pv = __expf(s[tt][j] - mnew);
            pw[r + tt * 16 + lo] = (__bf16)pv;           // C-layout -> LDS tile
        }
    }
    asm volatile("s_wait_dscnt 0" ::: "memory");         // wave-internal LDS sync

    // P repacked as two 16x32 A-fragments
    v16bf pA0 = load_frag2(pw + lo * 64 + hi * 16,      pw + lo * 64 + hi * 16 + 8);
    v16bf pA1 = load_frag2(pw + lo * 64 + 32 + hi * 16, pw + lo * 64 + 32 + hi * 16 + 8);

    // ---- row-sum via WMMA: rs = P * ones(64x16); every column holds the row sum ----
    v8f rs = {};
    rs = wmma_bf16(pA0, ones, rs);
    rs = wmma_bf16(pA1, ones, rs);
#pragma unroll
    for (int j = 0; j < 8; ++j) li[j] = li[j] * alpha[j] + rs[j];

    // ---- ctx = ctx*alpha + P (16x64) * V (64x64), V stored transposed [B,H,HD,S] ----
#pragma unroll
    for (int t = 0; t < 4; ++t) {
        const __bf16* vRow = Vt + ((size_t)bh * HDQ + t * 16 + lo) * SQ + kv0;
        v16bf vb0 = load_frag2(vRow + hi * 16,      vRow + hi * 16 + 8);
        v16bf vb1 = load_frag2(vRow + 32 + hi * 16, vRow + 32 + hi * 16 + 8);
#pragma unroll
        for (int j = 0; j < 8; ++j) cacc[t][j] *= alpha[j];
        cacc[t] = wmma_bf16(pA0, vb0, cacc[t]);
        cacc[t] = wmma_bf16(pA1, vb1, cacc[t]);
    }
}

// ---------------- Flash attention: one wave per (b,h, 16-query tile) ----------------
__global__ __launch_bounds__(256)
void flash_attn_wmma(const __bf16* __restrict__ Q, const __bf16* __restrict__ Km,
                     const __bf16* __restrict__ Vt, __bf16* __restrict__ ctx) {
    __shared__ alignas(16) __bf16 psh[8][16 * 64];   // per-wave 16x64 P staging (2 KB)

    const int lane = threadIdx.x & 31;
    const int wave = threadIdx.x >> 5;
    const int lo   = lane & 15;
    const int hi   = lane >> 4;
    const int w    = blockIdx.x * 8 + wave;
    const int qt   = w & (SQ / 16 - 1);      // 0..127
    const int bh   = w >> 7;                 // 0..63
    const int q0   = qt * 16;

    // Q fragments (16 x 64 split as two 16x32 A-frags); Q was pre-scaled by 1/sqrt(HD)
    const __bf16* qRow = Q + ((size_t)bh * SQ + q0 + lo) * HDQ;
    v16bf qa0 = load_frag2(qRow + hi * 16,      qRow + hi * 16 + 8);
    v16bf qa1 = load_frag2(qRow + 32 + hi * 16, qRow + 32 + hi * 16 + 8);

    // all-ones B fragment for WMMA row-sum
    v16bf ones;
#pragma unroll
    for (int i = 0; i < 16; ++i) ones[i] = (__bf16)1.0f;

    v8f   cacc[4] = {};
    float mi[8], li[8];
#pragma unroll
    for (int j = 0; j < 8; ++j) { mi[j] = -3.0e38f; li[j] = 0.0f; }

    const int nblk = (q0 + 79) >> 6;         // ceil((q0+16)/64) causal kv blocks of 64
    __bf16* pw = psh[wave];

    // steady state: mask-free blocks
    for (int ib = 0; ib < nblk - 1; ++ib)
        attn_kv_block(false, ib * 64, q0, lo, hi, bh, Km, Vt, pw,
                      qa0, qa1, ones, cacc, mi, li);
    // peeled final block: the only one that can cross the causal diagonal
    attn_kv_block(true, (nblk - 1) * 64, q0, lo, hi, bh, Km, Vt, pw,
                  qa0, qa1, ones, cacc, mi, li);

    // ---- normalize and write ctx as bf16 [B,S,D] (row-major for the O-projection) ----
    const int b = bh >> 4, h = bh & 15;
#pragma unroll
    for (int j = 0; j < 8; ++j) {
        float inv = 1.0f / li[j];
        size_t row = (size_t)b * SQ + q0 + j + 8 * hi;
#pragma unroll
        for (int t = 0; t < 4; ++t)
            ctx[row * DQ + h * HDQ + t * 16 + lo] = (__bf16)(cacc[t][j] * inv);
    }
}

// ---------------- host-side orchestration ----------------
extern "C" void kernel_launch(void* const* d_in, const int* in_sizes, int n_in,
                              void* d_out, int out_size, void* d_ws, size_t ws_size,
                              hipStream_t stream) {
    const float* x  = (const float*)d_in[0];
    const float* wq = (const float*)d_in[1];
    const float* wk = (const float*)d_in[2];
    const float* wv = (const float*)d_in[3];
    const float* wo = (const float*)d_in[4];
    float* out = (float*)d_out;

    char* ws = (char*)d_ws;
    const size_t szX = (size_t)MQ * DQ * 2;    // 16 MB bf16
    const size_t szW = (size_t)DQ * DQ * 2;    //  2 MB bf16
    __bf16* xbf  = (__bf16*)ws;              ws += szX;
    __bf16* wqT  = (__bf16*)ws;              ws += szW;
    __bf16* wkT  = (__bf16*)ws;              ws += szW;
    __bf16* wvT  = (__bf16*)ws;              ws += szW;
    __bf16* woT  = (__bf16*)ws;              ws += szW;
    __bf16* Qb   = (__bf16*)ws;              ws += szX;   // [B,H,S,HD] (pre-scaled)
    __bf16* Kb   = (__bf16*)ws;              ws += szX;   // [B,H,S,HD]
    __bf16* Vtb  = (__bf16*)ws;              ws += szX;   // [B,H,HD,S]
    __bf16* ctxb = (__bf16*)ws;              ws += szX;   // [B*S, D]

    // 1. cast x to bf16
    cvt_f32_bf16<<<(MQ * DQ) / 256, 256, 0, stream>>>(x, xbf, MQ * DQ);

    // 2. transpose + cast weights
    dim3 tg(DQ / 256, DQ);
    transpose_cvt<<<tg, 256, 0, stream>>>(wq, wqT);
    transpose_cvt<<<tg, 256, 0, stream>>>(wk, wkT);
    transpose_cvt<<<tg, 256, 0, stream>>>(wv, wvT);
    transpose_cvt<<<tg, 256, 0, stream>>>(wo, woT);

    // 3. QKV projections (WMMA GEMMs): Q (x 1/8) ,K -> [B,H,S,HD]; V -> [B,H,HD,S]
    dim3 gg(DQ / 64, MQ / 128);
    gemm_wmma_bf16<1><<<gg, 256, 0, stream>>>(xbf, wqT, nullptr, Qb,  MQ, DQ, DQ, 0.125f);
    gemm_wmma_bf16<1><<<gg, 256, 0, stream>>>(xbf, wkT, nullptr, Kb,  MQ, DQ, DQ, 1.0f);
    gemm_wmma_bf16<2><<<gg, 256, 0, stream>>>(xbf, wvT, nullptr, Vtb, MQ, DQ, DQ, 1.0f);

    // 4. causal flash attention (B*H*(S/16) = 8192 waves, 8 waves/block)
    flash_attn_wmma<<<(BQ * HQ * (SQ / 16)) / 8, 256, 0, stream>>>(Qb, Kb, Vtb, ctxb);

    // 5. output projection -> fp32 d_out
    gemm_wmma_bf16<0><<<gg, 256, 0, stream>>>(ctxb, woT, out, nullptr, MQ, DQ, DQ, 1.0f);
}
